// DeformableNCC_67001489817833
// MI455X (gfx1250) — compile-verified, matrix operand order checked by reading, not run on hardware
//
#include <hip/hip_runtime.h>
#include <hip/hip_bf16.h>

typedef _Float16 v16h __attribute__((ext_vector_type(16)));
typedef float    v8f  __attribute__((ext_vector_type(8)));
typedef float    v4f  __attribute__((ext_vector_type(4)));

#define B_    64
#define KCLS  32
#define T_    1280
#define C_    16
#define E_    64
#define CCH   32
#define POOL_ 10
#define PSEG  128          // T_/POOL_
#define FEAT  320          // CCH*POOL_
#define KCONV 80           // C_ * 5 taps
#define KPAD  96           // KCONV padded to 3 x 32

// ISA 7.12.2: 16-bit A 16x32 layout -> element e of lane-half h covers
// K = h*8 + e (e<8) and K = 16 + h*8 + (e-8) (e>=8).
__device__ __forceinline__ int kmap(int e, int h8) {
  return (e < 8) ? (h8 + e) : (16 + h8 + (e - 8));
}

// ---------------------------------------------------------------------------
// Kernel 1: Conv1d(16->32, k=5, pad=2) + bias + ReLU + avgpool(128) as
// implicit-GEMM on WMMA. One 256-thread block (8 waves) per (b, pool-bin):
//   M = 128 time steps (8 tiles), N = 32 channels (2 tiles), K = 80 -> 96.
// K ordering k = tap*16 + ci makes every 8-element fragment chunk a
// contiguous 16B slice of the f16 x tile in LDS (no im2col needed).
// Pool reduction via ds_add_f32 into a per-channel accumulator.
// ---------------------------------------------------------------------------
__global__ void __launch_bounds__(256)
k_conv_pool_wmma(const float* __restrict__ x,
                 const float* __restrict__ conv_w,
                 const float* __restrict__ conv_b,
                 float* __restrict__ pooled) {
  __shared__ _Float16 xs[(PSEG + 4) * C_];   // 132 rows x 16 ch, f16 (4224 B)
  __shared__ _Float16 Bsh[CCH * KPAD];       // 32 x 96, zero-padded (6144 B)
  __shared__ float    pool_acc[CCH];

  const int tid  = threadIdx.x;
  const int lane = tid & 31;
  const int wv   = tid >> 5;
  const int lm   = lane & 15;
  const int h8   = (lane >> 4) * 8;

  const int blk = blockIdx.x;          // b*10 + p
  const int p   = blk % POOL_;
  const int b   = blk / POOL_;

  // --- stage x tile (with halo, zero-padded at sequence ends) as f16 ---
  const int base_t = p * PSEG - 2;
  for (int i = tid; i < (PSEG + 4) * C_; i += 256) {
    const int row = i >> 4;
    const int c   = i & 15;
    const int tt  = base_t + row;
    xs[i] = (tt >= 0 && tt < T_) ? (_Float16)x[(b * T_ + tt) * C_ + c]
                                 : (_Float16)0.0f;
  }
  // --- stage weights as [n][k], k = tap*16 + ci, zero-padded to 96 ---
  for (int i = tid; i < CCH * KPAD; i += 256) {
    const int n = i / KPAD;
    const int k = i % KPAD;
    Bsh[i] = (k < KCONV) ? (_Float16)conv_w[n * KCONV + (k & 15) * 5 + (k >> 4)]
                         : (_Float16)0.0f;
  }
  if (tid < CCH) pool_acc[tid] = 0.0f;
  __syncthreads();

  // --- 16 tiles (8 mt x 2 nt); each wave owns 2 ---
  for (int i = 0; i < 2; ++i) {
    const int tile = wv * 2 + i;
    const int mt = tile >> 1, nt = tile & 1;
    const int m0 = mt * 16 + lm;       // time row within segment (A row)
    const int n  = nt * 16 + lm;       // output channel (B col)
    v8f c = {};
    for (int kb = 0; kb < 3; ++kb) {
      v16h a, bf;
      #pragma unroll
      for (int ch = 0; ch < 2; ++ch) {            // two 8-wide K chunks
        const int kstart = kb * 32 + ch * 16 + h8;
        const int j   = kstart >> 4;              // tap
        const int ci0 = kstart & 15;              // first input channel
        #pragma unroll
        for (int e = 0; e < 8; ++e) {
          const int d = ch * 8 + e;
          a[d]  = (kstart < KCONV) ? xs[(m0 + j) * C_ + ci0 + e]
                                   : (_Float16)0.0f;
          bf[d] = Bsh[n * KPAD + kstart + e];     // already zero-padded
        }
      }
      c = __builtin_amdgcn_wmma_f32_16x16x32_f16(false, a, false, bf,
                                                 (short)0, c, false, false);
    }
    // bias + ReLU + partial pool sum over this lane's 8 time rows
    const float bias = conv_b[n];
    float partial = 0.0f;
    #pragma unroll
    for (int r = 0; r < 8; ++r)
      partial += fmaxf(c[r] + bias, 0.0f);
    atomicAdd(&pool_acc[n], partial);             // ds_add_f32
  }
  __syncthreads();

  if (tid < CCH)
    pooled[b * FEAT + tid * POOL_ + p] = pool_acc[tid] * (1.0f / PSEG);
}

// ---------------------------------------------------------------------------
// Kernel 2: WMMA encoder GEMMs.
//   Phase A: embed[64,64] = relu(pooled[64,320] @ lin_w[64,320]^T + lin_b)
//   Phase B: warp[64,32]  = embed @ warp_W[32,64]^T + warp_b   (and off)
// Single 256-thread block = 8 waves; each wave owns 2 of 16 tiles per phase.
// ---------------------------------------------------------------------------
__global__ void __launch_bounds__(256)
k_encoder_wmma(const float* __restrict__ pooled,
               const float* __restrict__ lin_w,  const float* __restrict__ lin_b,
               const float* __restrict__ warp_W, const float* __restrict__ warp_b,
               const float* __restrict__ off_W,  const float* __restrict__ off_b,
               float* __restrict__ warp_out, float* __restrict__ off_out,
               float* __restrict__ out_tail /* d_out + B*K*T*C */) {
  __shared__ float embed[B_ * E_];   // 64x64 f32 = 16 KB

  const int tid  = threadIdx.x;
  const int lane = tid & 31;
  const int wv   = tid >> 5;
  const int lm   = lane & 15;
  const int h8   = (lane >> 4) * 8;

  // ---- Phase A: embed GEMM, M=N=64 (4x4 tiles), Kdim = 320 (10 steps) ----
  for (int i = 0; i < 2; ++i) {
    const int tile = wv * 2 + i;
    const int mt = tile >> 2, nt = tile & 3;
    v8f c = {};
    for (int kb = 0; kb < 10; ++kb) {
      v16h a, bf;
      #pragma unroll
      for (int e = 0; e < 16; ++e) {
        const int k = kb * 32 + kmap(e, h8);
        a[e]  = (_Float16)pooled[(mt * 16 + lm) * FEAT + k];
        bf[e] = (_Float16)lin_w [(nt * 16 + lm) * FEAT + k];
      }
      c = __builtin_amdgcn_wmma_f32_16x16x32_f16(false, a, false, bf,
                                                 (short)0, c, false, false);
    }
    const int n = nt * 16 + lm;
    const float bias = lin_b[n];
    #pragma unroll
    for (int r = 0; r < 8; ++r) {
      const int m = mt * 16 + h8 + r;
      embed[m * E_ + n] = fmaxf(c[r] + bias, 0.0f);
    }
  }
  __syncthreads();

  // ---- Phase B: warp/off heads, M=64, N=32 (4x2 tiles x 2 heads), K=64 ----
  for (int i = 0; i < 2; ++i) {
    const int task = wv * 2 + i;
    const int sel  = task >> 3;          // 0 = warp, 1 = off
    const int rem  = task & 7;
    const int mt = rem >> 1, nt = rem & 1;
    const float* Wm = sel ? off_W : warp_W;
    const float* bb = sel ? off_b : warp_b;
    v8f c = {};
    for (int kb = 0; kb < 2; ++kb) {
      v16h a, bf;
      #pragma unroll
      for (int e = 0; e < 16; ++e) {
        const int k = kb * 32 + kmap(e, h8);
        a[e]  = (_Float16)embed[(mt * 16 + lm) * E_ + k];
        bf[e] = (_Float16)Wm   [(nt * 16 + lm) * E_ + k];
      }
      c = __builtin_amdgcn_wmma_f32_16x16x32_f16(false, a, false, bf,
                                                 (short)0, c, false, false);
    }
    const int n = nt * 16 + lm;
    const float bias = bb[n];
    float* dst = sel ? off_out : warp_out;
    #pragma unroll
    for (int r = 0; r < 8; ++r) {
      const int m = mt * 16 + h8 + r;
      const float v = c[r] + bias;
      dst[m * KCLS + n] = v;
      out_tail[sel * (B_ * KCLS) + m * KCLS + n] = v;
    }
  }
}

// ---------------------------------------------------------------------------
// Kernel 3: dominant streaming stage (167.8 MB out). 4 lanes per (b,k,t) row,
// float4 prototype gathers (L2-resident, ~2.6 MB reused 64x) + non-temporal
// float4 stores so the output stream doesn't evict the prototype set from L2.
// ---------------------------------------------------------------------------
__global__ void __launch_bounds__(256)
k_warp_stream(const float* __restrict__ proto,
              const float* __restrict__ warp_v,
              const float* __restrict__ off_v,
              float* __restrict__ out) {
  const int g  = blockIdx.x * 256 + threadIdx.x;
  const int c4 = g & 3;
  const int r  = g >> 2;               // row id: (b*32+k)*1280 + t
  const int t  = r % T_;
  const int bk = r / T_;
  const int k  = bk & (KCLS - 1);

  const float wv = warp_v[bk];
  const float ov = off_v[bk];

  float idx = fminf(fmaxf((float)t - wv, 0.0f), (float)(T_ - 1));
  const float f0 = floorf(idx);
  int i0 = (int)f0;
  int i1 = i0 + 1; if (i1 > T_ - 1) i1 = T_ - 1;
  const float fr = idx - f0;

  const v4f p0 = *(const v4f*)(proto + ((k * T_ + i0) * C_ + c4 * 4));
  const v4f p1 = *(const v4f*)(proto + ((k * T_ + i1) * C_ + c4 * 4));
  v4f o;
  #pragma unroll
  for (int e = 0; e < 4; ++e)
    o[e] = fmaf(fr, p1[e] - p0[e], p0[e]) + ov;   // lerp(p0,p1) + off

  __builtin_nontemporal_store(o, (v4f*)(out + (size_t)r * C_ + (size_t)(c4 * 4)));
}

// ---------------------------------------------------------------------------
extern "C" void kernel_launch(void* const* d_in, const int* in_sizes, int n_in,
                              void* d_out, int out_size, void* d_ws, size_t ws_size,
                              hipStream_t stream) {
  const float* x      = (const float*)d_in[0];
  const float* proto  = (const float*)d_in[1];
  const float* conv_w = (const float*)d_in[2];
  const float* conv_b = (const float*)d_in[3];
  const float* lin_w  = (const float*)d_in[4];
  const float* lin_b  = (const float*)d_in[5];
  const float* warp_W = (const float*)d_in[6];
  const float* warp_b = (const float*)d_in[7];
  const float* off_W  = (const float*)d_in[8];
  const float* off_b  = (const float*)d_in[9];
  float* out = (float*)d_out;

  float* pooled   = (float*)d_ws;                 //  64*320 f32
  float* warp_buf = pooled + B_ * FEAT;           //  64*32  f32
  float* off_buf  = warp_buf + B_ * KCLS;         //  64*32  f32

  const size_t warped_elems = (size_t)B_ * KCLS * T_ * C_;   // 41,943,040

  k_conv_pool_wmma<<<B_ * POOL_, 256, 0, stream>>>(x, conv_w, conv_b, pooled);

  k_encoder_wmma<<<1, 256, 0, stream>>>(pooled, lin_w, lin_b,
                                        warp_W, warp_b, off_W, off_b,
                                        warp_buf, off_buf, out + warped_elems);

  const int total = B_ * KCLS * T_ * 4;           // 10,485,760 lanes
  k_warp_stream<<<total / 256, 256, 0, stream>>>(proto, warp_buf, off_buf, out);
}
